// S5_word_27685359190749
// MI455X (gfx1250) — matrix operand order repaired and checked by reading, not compiled, not run on hardware
//
#include <hip/hip_runtime.h>
#include <stdint.h>

#define BATCH  256
#define TLEN   32768
#define NPERM  120
#define DDIM   5
#define CHUNK  1024
#define NCHUNK (TLEN / CHUNK)   // 32 chunks per batch
#define SUPER  (CHUNK / 32)     // 32 super-steps of 32 tokens per chunk

typedef __attribute__((ext_vector_type(16))) _Float16 v16h;
typedef __attribute__((ext_vector_type(8)))  float    v8f;

// workspace layout (bytes)
#define WS_ENC   0        // int32[3125]   base5-encode -> perm index
#define WS_DIG   16384    // uint8[120*5]  perm digits p[i][r]
#define WS_COMP  17408    // uint8[120*120] composition table (built by WMMA)
#define WS_TOT   32768    // int32[BATCH*NCHUNK] chunk totals
#define WS_PRE   65536    // int32[BATCH*NCHUNK] exclusive chunk prefixes

// ---------------------------------------------------------------------------
// Kernel 0: extract permutation digits + encode->index table from perm_mats.
// (P x)[r] = x[p[r]] where mats[i][r][p[r]] == 1.
// ---------------------------------------------------------------------------
__global__ void k_digits(const float* __restrict__ mats,
                         int* __restrict__ enc_to_idx,
                         uint8_t* __restrict__ dig) {
    int i = threadIdx.x;
    if (i >= NPERM) return;
    int enc = 0, p5 = 1;
    for (int r = 0; r < DDIM; ++r) {
        int c = 0;
        for (int j = 0; j < DDIM; ++j)
            if (mats[i * 25 + r * 5 + j] > 0.5f) c = j;
        dig[i * 5 + r] = (uint8_t)c;
        enc += c * p5;
        p5 *= 5;
    }
    enc_to_idx[enc] = i;
}

// ---------------------------------------------------------------------------
// Kernel 1: build comp[a][b] = index(P[a] @ P[b]) with real WMMA.
// One wave per (a,b) pair; 5x5 product embedded in a 16x16x32 f16 tile.
// Grid 1800 blocks x 256 threads = 14400 waves exactly (no divergence: EXEC
// stays all-ones around the WMMA).
// ---------------------------------------------------------------------------
__global__ void k_compose(const float* __restrict__ mats,
                          const int* __restrict__ enc_to_idx,
                          uint8_t* __restrict__ comp) {
    int wave = (blockIdx.x * blockDim.x + threadIdx.x) >> 5;
    int lane = threadIdx.x & 31;
    int a = wave / NPERM;
    int b = wave % NPERM;
    const float* Am = mats + a * 25;
    const float* Bm = mats + b * 25;

    int h = lane >> 4;       // half of wave
    int m = lane & 15;       // A row / D column group
    int n = lane & 15;       // B column

    v16h af{}, bf{};
#pragma unroll
    for (int e = 0; e < 16; ++e) {
        int j = e >> 1, bit = e & 1;
        // A 16x32 f16 layout: V0..3 -> K = h*8 + 2j (+bit); V4..7 -> K = 16 + h*8 + 2(j-4) (+bit)
        int ka = (j < 4 ? h * 8 + 2 * j : 16 + h * 8 + 2 * (j - 4)) + bit;
        float av = (m < 5 && ka < 5) ? Am[m * 5 + ka] : 0.0f;
        af[e] = (_Float16)av;
        // B 32x16 f16 layout: same K order as A: e<8 -> K = h*8 + e; e>=8 -> K = 16 + h*8 + (e-8)
        int kb = (e < 8 ? h * 8 + e : 16 + h * 8 + (e - 8));
        float bv = (kb < 5 && n < 5) ? Bm[kb * 5 + n] : 0.0f;
        bf[e] = (_Float16)bv;
    }

    v8f c{};
    c = __builtin_amdgcn_wmma_f32_16x16x32_f16(false, af, false, bf,
                                               (short)0, c, false, false);

    // D layout: VGPR r holds row M=r (lanes 0-15, N=lane) / M=8+r (lanes 16-31).
    // Rows 0..4 live entirely in lanes 0-15; find the '1' column per row via ballot.
    int enc = 0, p5 = 1;
#pragma unroll
    for (int r = 0; r < 5; ++r) {
        float v = (h == 0) ? c[r] : 0.0f;
        unsigned long long mask = __ballot(v > 0.5f);
        int col = __ffsll(mask) - 1;
        enc += col * p5;
        p5 *= 5;
    }
    if (lane == 0) comp[a * NPERM + b] = (uint8_t)enc_to_idx[enc];
}

// ---------------------------------------------------------------------------
// Wave32 inclusive scan of group elements: v_l = u_l o u_{l-1} o ... o u_0
// with o(x,y) = comp[x][y] = index(P[x] @ P[y])  (newer element on the left).
// ---------------------------------------------------------------------------
__device__ __forceinline__ int wave_scan(int v, int lane, const uint8_t* comp) {
#pragma unroll
    for (int d = 1; d < 32; d <<= 1) {
        int w = __shfl_up(v, (unsigned)d, 32);
        if (lane >= d) v = (int)comp[v * NPERM + w];
    }
    return v;
}

// ---------------------------------------------------------------------------
// Kernel 2: one wave per (batch, chunk) -> total group element of the chunk.
// ---------------------------------------------------------------------------
__global__ void k_chunktot(const int* __restrict__ tokens,
                           const uint8_t* __restrict__ comp_g,
                           int* __restrict__ tot) {
    __shared__ uint8_t comp[NPERM * NPERM];
    for (int i = threadIdx.x; i < NPERM * NPERM; i += blockDim.x)
        comp[i] = comp_g[i];
    __syncthreads();

    int lane  = threadIdx.x & 31;
    int wv    = threadIdx.x >> 5;              // 0..7
    int bIdx  = blockIdx.x >> 2;               // 4 blocks per batch
    int chunk = ((blockIdx.x & 3) << 3) + wv;  // 0..31

    const int* tk = tokens + (size_t)bIdx * TLEN + (size_t)chunk * CHUNK;
    int g = 0;  // identity (itertools.permutations yields (0,1,2,3,4) first)
    for (int s = 0; s < SUPER; ++s) {
        int v = tk[s * 32 + lane];
        v = wave_scan(v, lane, comp);
        int st = __shfl(v, 31, 32);            // total of these 32 tokens
        g = (int)comp[st * NPERM + g];         // newer super-step on the left
    }
    if (lane == 0) tot[bIdx * NCHUNK + chunk] = g;
}

// ---------------------------------------------------------------------------
// Kernel 3: per-batch exclusive prefix over the 32 chunk totals.
// ---------------------------------------------------------------------------
__global__ void k_prefix(const int* __restrict__ tot,
                         const uint8_t* __restrict__ comp,
                         int* __restrict__ pre) {
    int b = threadIdx.x;
    if (b >= BATCH) return;
    int g = 0;
    for (int j = 0; j < NCHUNK; ++j) {
        pre[b * NCHUNK + j] = g;
        g = (int)comp[tot[b * NCHUNK + j] * NPERM + g];
    }
}

// ---------------------------------------------------------------------------
// Kernel 4: emit. One wave per (batch, chunk); per super-step: coalesced token
// load, 5-round wave scan, compose with chunk prefix, store 5 floats/lane from
// the per-batch LDS table of pre-permuted x0 vectors (only 120 possible).
// ---------------------------------------------------------------------------
__global__ void k_emit(const int* __restrict__ tokens,
                       const float* __restrict__ x0,
                       const uint8_t* __restrict__ comp_g,
                       const uint8_t* __restrict__ dig,
                       const int* __restrict__ pre,
                       float* __restrict__ out) {
    __shared__ uint8_t comp[NPERM * NPERM];
    __shared__ float vec[NPERM * DDIM];

    int bIdx = blockIdx.x >> 2;
    for (int i = threadIdx.x; i < NPERM * NPERM; i += blockDim.x)
        comp[i] = comp_g[i];
    if (threadIdx.x < NPERM) {
        for (int r = 0; r < DDIM; ++r)
            vec[threadIdx.x * DDIM + r] =
                x0[bIdx * DDIM + dig[threadIdx.x * 5 + r]];
    }
    __syncthreads();

    int lane  = threadIdx.x & 31;
    int wv    = threadIdx.x >> 5;
    int chunk = ((blockIdx.x & 3) << 3) + wv;

    const int* tk = tokens + (size_t)bIdx * TLEN + (size_t)chunk * CHUNK;
    float* ob = out + ((size_t)bIdx * TLEN + (size_t)chunk * CHUNK) * DDIM;
    int gbase = pre[bIdx * NCHUNK + chunk];

    for (int s = 0; s < SUPER; ++s) {
        int v = tk[s * 32 + lane];
        v = wave_scan(v, lane, comp);                 // prefix within 32 tokens
        int g = (int)comp[v * NPERM + gbase];         // global inclusive prefix
        float* o = ob + (size_t)(s * 32 + lane) * DDIM;
#pragma unroll
        for (int r = 0; r < DDIM; ++r) o[r] = vec[g * DDIM + r];
        gbase = __shfl(g, 31, 32);                    // carry to next super-step
    }
}

// ---------------------------------------------------------------------------
extern "C" void kernel_launch(void* const* d_in, const int* in_sizes, int n_in,
                              void* d_out, int out_size, void* d_ws, size_t ws_size,
                              hipStream_t stream) {
    (void)in_sizes; (void)n_in; (void)out_size; (void)ws_size;
    const float* x0     = (const float*)d_in[0];   // (B, 5)
    const int*   tokens = (const int*)d_in[1];     // (B, T)
    const float* mats   = (const float*)d_in[2];   // (120, 5, 5)
    float* out = (float*)d_out;                    // (B, T, 5)

    uint8_t* ws  = (uint8_t*)d_ws;
    int*     enc = (int*)(ws + WS_ENC);
    uint8_t* dig = ws + WS_DIG;
    uint8_t* cmp = ws + WS_COMP;
    int*     tot = (int*)(ws + WS_TOT);
    int*     pre = (int*)(ws + WS_PRE);

    k_digits  <<<1,    128, 0, stream>>>(mats, enc, dig);
    k_compose <<<1800, 256, 0, stream>>>(mats, enc, cmp);   // 14400 waves, WMMA
    k_chunktot<<<BATCH * 4, 256, 0, stream>>>(tokens, cmp, tot);
    k_prefix  <<<1,    256, 0, stream>>>(tot, cmp, pre);
    k_emit    <<<BATCH * 4, 256, 0, stream>>>(tokens, x0, cmp, dig, pre, out);
}
